// networkarch_16690242912804
// MI455X (gfx1250) — matrix operand order repaired and checked by reading, not compiled, not run on hardware
//
#include <hip/hip_runtime.h>

// CDNA5 / gfx1250, wave32. Core math: v_wmma_f32_16x16x32_f16 computing the
// TRANSPOSED product Y^T = W^T x X^T so that:
//   A-frag = f16-transposed weights Wt[n][k]   (global, L2-resident)
//   B-frag = row-major f16 activations          (LDS, 2 x 8 contiguous K / lane)
//   C/D    = each lane holds 8 CONTIGUOUS n of one row m -> fully vectorized
//            bias / relu / residual / store epilogues.

typedef __attribute__((ext_vector_type(16))) _Float16 v16h;
typedef __attribute__((ext_vector_type(8)))  _Float16 v8h;
typedef __attribute__((ext_vector_type(4)))  _Float16 v4h;
typedef __attribute__((ext_vector_type(8)))  float    v8f;

#define N_IN      128
#define LDS_RS    136   // LDS activation row stride in halfs
#define STG_RS    132   // f32 output staging row stride in floats
#define ROWS_PER_BLOCK 128

__device__ __forceinline__ v16h cat8(v8h lo, v8h hi) {
  return __builtin_shufflevector(lo, hi, 0,1,2,3,4,5,6,7,8,9,10,11,12,13,14,15);
}

// One 16x16 output tile (rows m = wave's 16, cols n = nt*16..+15), K = NKS*32.
// actw: wave's LDS activation base (row-major, stride LDS_RS)
// wt:   f16 transposed weights [128][klen] in global
template<int NKS>
__device__ __forceinline__ v8f gemm_tile(const _Float16* actw,
                                         const _Float16* __restrict__ wt,
                                         int klen, int lane, int nt) {
  const int m     = lane & 15;
  const int khalf = lane >> 4;
  const _Float16* arow = actw + m * LDS_RS + khalf * 8;             // B source
  const _Float16* wrow = wt + (size_t)(nt * 16 + m) * klen + khalf * 8; // A source
  v8f acc = {0.f,0.f,0.f,0.f,0.f,0.f,0.f,0.f};
  #pragma unroll
  for (int ks = 0; ks < NKS; ++ks) {
    v8h wlo = *(const v8h*)(wrow + ks * 32);
    v8h whi = *(const v8h*)(wrow + ks * 32 + 16);
    v8h alo = *(const v8h*)(arow + ks * 32);
    v8h ahi = *(const v8h*)(arow + ks * 32 + 16);
    acc = __builtin_amdgcn_wmma_f32_16x16x32_f16(false, cat8(wlo, whi),
                                                 false, cat8(alo, ahi),
                                                 (short)0, acc, false, false);
  }
  return acc;
}

// Generic fused DenseResBlock (see round-0 analysis for the algebra):
//   t   = (PREK ? x @ preW : x)     preW f16-transposed, K padded to PREK
//   h   = relu(t @ Wi + bi), i=0..3 ; out = t + h @ Wo + bo
// Output row r -> out + (r % out_mod)*out_smod + (r / out_mod)*out_sdiv.
// 256 threads = 8 waves; each wave owns 16 rows -> barrier-free.
template<int PREK>
__global__ __launch_bounds__(256) void resblock_kernel(
    const float* __restrict__ x, int x_width, int x_stride,
    const _Float16* __restrict__ preWt,
    const _Float16* __restrict__ Wt,      // [5][128][128] f16, transposed
    const float* __restrict__ bh,         // [4][128]
    const float* __restrict__ bo,         // [128]
    float* __restrict__ out,
    int out_mod, long long out_smod, long long out_sdiv)
{
  extern __shared__ __align__(16) unsigned char smem[];
  _Float16* bufA  = (_Float16*)smem;                          // ping / raw staging
  _Float16* bufB  = bufA + ROWS_PER_BLOCK * LDS_RS;           // pong
  _Float16* bufX  = bufB + ROWS_PER_BLOCK * LDS_RS;           // input / residual
  float*    stage = (float*)(bufX + ROWS_PER_BLOCK * LDS_RS); // f32 staging

  const int lane    = threadIdx.x & 31;
  const int wave    = threadIdx.x >> 5;
  const int rowbase = blockIdx.x * ROWS_PER_BLOCK + wave * 16;

  _Float16* rawb  = (PREK ? bufA : bufX) + wave * 16 * LDS_RS;
  _Float16* xw    = bufX + wave * 16 * LDS_RS;
  _Float16* pingw = bufA + wave * 16 * LDS_RS;
  _Float16* pongw = bufB + wave * 16 * LDS_RS;
  float*    stgw  = stage + wave * 16 * STG_RS;

  // ---- load this wave's 16 raw rows (f32 -> f16), zero-pad to KP ----
  const int KP = PREK ? PREK : N_IN;
  if (x_width == N_IN) {                       // vector path: 1 float4/lane/row
    for (int r = 0; r < 16; ++r) {
      float4 v4 = *(const float4*)(x + (size_t)(rowbase + r) * x_stride + lane * 4);
      v4h h4; h4[0] = (_Float16)v4.x; h4[1] = (_Float16)v4.y;
              h4[2] = (_Float16)v4.z; h4[3] = (_Float16)v4.w;
      *(v4h*)(rawb + r * LDS_RS + lane * 4) = h4;
    }
  } else {                                     // narrow path (width 21 -> 32)
    for (int r = 0; r < 16; ++r) {
      const float* xr = x + (size_t)(rowbase + r) * x_stride;
      for (int k = lane; k < KP; k += 32)
        rawb[r * LDS_RS + k] = (_Float16)((k < x_width) ? xr[k] : 0.0f);
    }
  }

  // ---- optional fused pre-multiply: bufX = raw @ preW ----
  if (PREK) {
    #pragma unroll
    for (int nt = 0; nt < 8; ++nt) {
      v8f acc = gemm_tile<PREK / 32>(rawb, preWt, PREK, lane, nt);
      const int m  = lane & 15;
      const int nb = nt * 16 + (lane >> 4) * 8;
      v8h h8;
      #pragma unroll
      for (int e = 0; e < 8; ++e) h8[e] = (_Float16)acc[e];
      *(v8h*)(xw + m * LDS_RS + nb) = h8;
    }
  }

  // ---- 4 hidden relu layers ----
  const _Float16* act = xw;
  _Float16* nxt = pingw;
  #pragma unroll
  for (int layer = 0; layer < 4; ++layer) {
    const _Float16* wt = Wt + (size_t)layer * N_IN * N_IN;
    const float* bias  = bh + layer * N_IN;
    #pragma unroll
    for (int nt = 0; nt < 8; ++nt) {
      v8f acc = gemm_tile<4>(act, wt, N_IN, lane, nt);
      const int m  = lane & 15;
      const int nb = nt * 16 + (lane >> 4) * 8;
      float4 b0 = *(const float4*)(bias + nb);
      float4 b1 = *(const float4*)(bias + nb + 4);
      const float bb[8] = {b0.x, b0.y, b0.z, b0.w, b1.x, b1.y, b1.z, b1.w};
      v8h h8;
      #pragma unroll
      for (int e = 0; e < 8; ++e)
        h8[e] = (_Float16)fmaxf(acc[e] + bb[e], 0.0f);
      *(v8h*)(nxt + m * LDS_RS + nb) = h8;
    }
    act = nxt;
    nxt = (layer & 1) ? pingw : pongw;
  }

  // ---- output layer: residual + bias, f32 to staging ----
  {
    const _Float16* wt = Wt + (size_t)4 * N_IN * N_IN;
    #pragma unroll
    for (int nt = 0; nt < 8; ++nt) {
      v8f acc = gemm_tile<4>(act, wt, N_IN, lane, nt);
      const int m  = lane & 15;
      const int nb = nt * 16 + (lane >> 4) * 8;
      float4 b0 = *(const float4*)(bo + nb);
      float4 b1 = *(const float4*)(bo + nb + 4);
      const float bb[8] = {b0.x, b0.y, b0.z, b0.w, b1.x, b1.y, b1.z, b1.w};
      v8h r8 = *(const v8h*)(xw + m * LDS_RS + nb);     // residual skip
      float c[8];
      #pragma unroll
      for (int e = 0; e < 8; ++e) c[e] = acc[e] + bb[e] + (float)r8[e];
      *(float4*)(stgw + m * STG_RS + nb)     = make_float4(c[0], c[1], c[2], c[3]);
      *(float4*)(stgw + m * STG_RS + nb + 4) = make_float4(c[4], c[5], c[6], c[7]);
    }
  }

  // ---- coalesced writeback: 16 rows x 128 f32 (32 lanes x float4 each) ----
  for (int r = 0; r < 16; ++r) {
    const int gr = rowbase + r;
    float* orow = out + (long long)(gr % out_mod) * out_smod
                      + (long long)(gr / out_mod) * out_sdiv;
    float4 v4 = *(const float4*)(stgw + r * STG_RS + lane * 4);
    *(float4*)(orow + lane * 4) = v4;
  }
}

// ---------------------------------------------------------------------------
// Weight prep: f32 [nmat][K][N] -> f16 transposed [nmat][N][K]
__global__ void transpose_f16_kernel(const float* __restrict__ src,
                                     _Float16* __restrict__ dst, int nmat) {
  int idx = blockIdx.x * blockDim.x + threadIdx.x;
  if (idx >= nmat * 16384) return;
  int m = idx >> 14, rem = idx & 16383;
  int n = rem >> 7, k = rem & 127;
  dst[idx] = (_Float16)src[m * 16384 + k * 128 + n];
}

// MT[n][k] = (ienc_W @ idec_W)[k][n]  -- rank-21 bottleneck collapsed to 128x128
__global__ void latentm_kernel(const float* __restrict__ ienc,
                               const float* __restrict__ idec,
                               _Float16* __restrict__ MT) {
  int idx = blockIdx.x * blockDim.x + threadIdx.x;
  if (idx >= 16384) return;
  int n = idx >> 7, k = idx & 127;
  float s = 0.f;
  for (int t = 0; t < 21; ++t) s += ienc[k * 21 + t] * idec[t * 128 + n];
  MT[idx] = (_Float16)s;
}

// idecT[n][k] = idec_W[k][n], K padded 21 -> 32 with zeros
__global__ void idect_kernel(const float* __restrict__ idec, _Float16* __restrict__ dT) {
  int idx = blockIdx.x * blockDim.x + threadIdx.x;
  if (idx >= 128 * 32) return;
  int n = idx >> 5, k = idx & 31;
  dT[idx] = (_Float16)((k < 21) ? idec[k * 128 + n] : 0.0f);
}

// Latent rollout in f32 VALU (~45 MFLOP, precision-critical, trivially cheap).
__global__ __launch_bounds__(128) void rollout_kernel(const float* __restrict__ pe,
                                                      const float* __restrict__ ienc,
                                                      const float* __restrict__ L,
                                                      float* __restrict__ zs) {
  __shared__ float sL[21 * 21];
  __shared__ float sE[128 * 21];
  for (int i = threadIdx.x; i < 441;  i += 128) sL[i] = L[i];
  for (int i = threadIdx.x; i < 2688; i += 128) sE[i] = ienc[i];
  __syncthreads();
  const int b = blockIdx.x * 128 + threadIdx.x;     // grid = 16 -> b in [0,2048)
  const float* per = pe + (size_t)b * 51 * 128;
  float z[21];
  #pragma unroll
  for (int j = 0; j < 21; ++j) z[j] = 0.f;
  for (int i = 0; i < 128; ++i) {
    const float xv = per[i];
    #pragma unroll
    for (int j = 0; j < 21; ++j) z[j] += xv * sE[i * 21 + j];
  }
  for (int s = 0; s < 50; ++s) {
    float zn[21];
    #pragma unroll
    for (int j = 0; j < 21; ++j) {
      float a = 0.f;
      #pragma unroll
      for (int i = 0; i < 21; ++i) a += z[i] * sL[i * 21 + j];
      zn[j] = a;
    }
    float* zrow = zs + ((size_t)s * 2048 + b) * 21;
    #pragma unroll
    for (int j = 0; j < 21; ++j) { z[j] = zn[j]; zrow[j] = zn[j]; }
  }
}

// ---------------------------------------------------------------------------
extern "C" void kernel_launch(void* const* d_in, const int* in_sizes, int n_in,
                              void* d_out, int out_size, void* d_ws, size_t ws_size,
                              hipStream_t stream) {
  (void)in_sizes; (void)n_in; (void)out_size; (void)ws_size;
  const float* inputs = (const float*)d_in[0];
  const float* L      = (const float*)d_in[1];
  const float* ienc   = (const float*)d_in[2];
  const float* idec   = (const float*)d_in[3];
  const float* encW   = (const float*)d_in[4];
  const float* encB   = (const float*)d_in[5];
  const float* encWo  = (const float*)d_in[6];
  const float* encBo  = (const float*)d_in[7];
  const float* decW   = (const float*)d_in[8];
  const float* decB   = (const float*)d_in[9];
  const float* decWo  = (const float*)d_in[10];
  const float* decBo  = (const float*)d_in[11];

  // workspace carve (~62.4 MB total)
  unsigned char* ws = (unsigned char*)d_ws;
  size_t off = 0;
  auto carve = [&](size_t bytes) -> void* {
    void* p = ws + off; off = (off + bytes + 255) & ~(size_t)255; return p;
  };
  _Float16* encWt = (_Float16*)carve(5 * 16384 * sizeof(_Float16));
  _Float16* decWt = (_Float16*)carve(5 * 16384 * sizeof(_Float16));
  _Float16* MT    = (_Float16*)carve(16384 * sizeof(_Float16));
  _Float16* idT   = (_Float16*)carve(128 * 32 * sizeof(_Float16));
  float*    pe    = (float*)carve((size_t)104448 * 128 * sizeof(float));
  float*    zs    = (float*)carve((size_t)50 * 2048 * 21 * sizeof(float));

  // weight prep (deterministic, inputs never mutated)
  transpose_f16_kernel<<<(4 * 16384 + 255) / 256, 256, 0, stream>>>(encW, encWt, 4);
  transpose_f16_kernel<<<(16384 + 255) / 256, 256, 0, stream>>>(encWo, encWt + 4 * 16384, 1);
  transpose_f16_kernel<<<(4 * 16384 + 255) / 256, 256, 0, stream>>>(decW, decWt, 4);
  transpose_f16_kernel<<<(16384 + 255) / 256, 256, 0, stream>>>(decWo, decWt + 4 * 16384, 1);
  latentm_kernel<<<16384 / 256, 256, 0, stream>>>(ienc, idec, MT);
  idect_kernel<<<(128 * 32) / 256, 256, 0, stream>>>(idec, idT);

  const int SMEM = 3 * ROWS_PER_BLOCK * LDS_RS * (int)sizeof(_Float16)
                 + ROWS_PER_BLOCK * STG_RS * (int)sizeof(float);      // 172032 B
  hipFuncSetAttribute(reinterpret_cast<const void*>(&resblock_kernel<0>),
                      hipFuncAttributeMaxDynamicSharedMemorySize, SMEM);
  hipFuncSetAttribute(reinterpret_cast<const void*>(&resblock_kernel<32>),
                      hipFuncAttributeMaxDynamicSharedMemorySize, SMEM);
  hipFuncSetAttribute(reinterpret_cast<const void*>(&resblock_kernel<128>),
                      hipFuncAttributeMaxDynamicSharedMemorySize, SMEM);

  float* out1 = (float*)d_out;                       // autoencoder_output [104448,128]
  float* out2 = out1 + (size_t)104448 * 128;         // outer_auto_output  [104448,128]
  float* out3 = out2 + (size_t)104448 * 128;         // predictions [2048,50,128]

  // pe = resblock_enc(inputs)            104448 rows = 816 blocks
  resblock_kernel<0><<<816, 256, SMEM, stream>>>(inputs, 128, 128, nullptr,
                                                 encWt, encB, encBo, pe, 1, 0, 128);
  // latent rollout (needs pe rows b*51)
  rollout_kernel<<<16, 128, 0, stream>>>(pe, ienc, L, zs);
  // autoencoder_output = resblock_dec(pe @ M)
  resblock_kernel<128><<<816, 256, SMEM, stream>>>(pe, 128, 128, MT,
                                                   decWt, decB, decBo, out1, 1, 0, 128);
  // outer_auto_output = resblock_dec(pe)
  resblock_kernel<0><<<816, 256, SMEM, stream>>>(pe, 128, 128, nullptr,
                                                 decWt, decB, decBo, out2, 1, 0, 128);
  // predictions = resblock_dec(zs @ idec), row (s*2048+b) -> out3[b*6400 + s*128]
  resblock_kernel<32><<<800, 256, SMEM, stream>>>(zs, 21, 21, idT,
                                                  decWt, decB, decBo, out3, 2048, 6400, 128);
}
